// RoIHead_8297876815897
// MI455X (gfx1250) — matrix operand order repaired and checked by reading, not compiled, not run on hardware
//
#include <hip/hip_runtime.h>
#include <hip/hip_bf16.h>
#include <math.h>

// ---------------------------------------------------------------------------
// Problem constants (from reference)
// ---------------------------------------------------------------------------
#define BB    4
#define CC    128
#define HH    50
#define WW    50
#define RR    512
#define OUTHW 7
#define POOLK (CC * OUTHW * OUTHW)   // 6272
#define FC    1024
#define NCLS  21
#define NLOC  (NCLS * 4)             // 84
#define SCALEF 0.0625f

#define KB1   (POOLK / 32)           // 196 K-blocks for fc1
#define KB2   (FC / 32)              // 32  K-blocks for fc2/heads
#define NB_FC (FC / 16)              // 64  N-frags for fc outputs
#define NB_LOC 6                     // ceil(84/16)
#define NB_SCR 2                     // ceil(21/16)

typedef __attribute__((ext_vector_type(16))) __bf16 v16bf;
typedef __attribute__((ext_vector_type(8)))  float  v8f;
typedef unsigned short ushort_t;

// ---------------------------------------------------------------------------
// bf16x3 split into a packed hi/lo buffer: fragment chunk per lane is
// [16 x hi bf16][16 x lo bf16] = 64 bytes. idx addresses the hi slot.
// ---------------------------------------------------------------------------
static __device__ __forceinline__ void split_store(float v,
                                                   ushort_t* __restrict__ P,
                                                   size_t idx) {
    unsigned u  = __float_as_uint(v);
    unsigned hb = (u + 0x7FFFu + ((u >> 16) & 1u)) >> 16;
    float    fh = __uint_as_float(hb << 16);
    float    lo = v - fh;
    unsigned ul = __float_as_uint(lo);
    unsigned lb = (ul + 0x7FFFu + ((ul >> 16) & 1u)) >> 16;
    P[idx]      = (ushort_t)hb;
    P[idx + 16] = (ushort_t)lb;
}

// A-operand swizzle: element (m,k) of an [M,K] matrix -> hi-slot index in the
// packed fragment buffer (16x32 A frag, contiguous k = 16*lanehi + e).
// frag(mb,kb) occupies 32 lanes x 32 ushorts at ((mb*KB+kb)*32+lane)*32.
static __device__ __forceinline__ size_t aswz_idx(int m, int k, int KB) {
    int mb = m >> 4, kb = k >> 5, kk = k & 31;
    int lanehi = (kk >> 4) & 1;                      // K in {16..31} -> lanes 16..31
    int e = kk & 15;                                 // element slot 0..15
    int lane = lanehi * 16 + (m & 15);
    return ((size_t)(mb * KB + kb) * 32 + lane) * 32 + e;
}

// ---------------------------------------------------------------------------
// Kernel 1: RoI max pool -> packed swizzled bf16 hi/lo A operand for fc1.
// (Reference permutes roi columns to (idx, y1, x1, y2, x2) before pooling.)
// ---------------------------------------------------------------------------
__global__ __launch_bounds__(256)
void roi_pool_kernel(const float* __restrict__ x,
                     const float* __restrict__ rois,
                     const int*   __restrict__ ridx,
                     ushort_t* __restrict__ Apk) {
    int r = blockIdx.x;
    int b = ridx[r];
    float c0 = rois[r * 4 + 0];
    float c1 = rois[r * 4 + 1];
    float c2 = rois[r * 4 + 2];
    float c3 = rois[r * 4 + 3];
    int xs = (int)floorf(c1 * SCALEF + 0.5f);
    int ys = (int)floorf(c0 * SCALEF + 0.5f);
    int xe = (int)floorf(c3 * SCALEF + 0.5f);
    int ye = (int)floorf(c2 * SCALEF + 0.5f);
    float roiw = fmaxf((float)(xe - xs + 1), 1.0f);
    float roih = fmaxf((float)(ye - ys + 1), 1.0f);
    float bw = roiw / (float)OUTHW;
    float bh = roih / (float)OUTHW;
    const float* xb = x + (size_t)b * CC * HH * WW;

    for (int e = threadIdx.x; e < POOLK; e += 256) {
        int c  = e / 49;
        int rm = e - c * 49;
        int ph = rm / 7;
        int pw = rm - ph * 7;
        int hs = min(max((int)floorf((float)ph * bh) + ys, 0), HH);
        int he = min(max((int)ceilf(((float)ph + 1.0f) * bh) + ys, 0), HH);
        int ws = min(max((int)floorf((float)pw * bw) + xs, 0), WW);
        int we = min(max((int)ceilf(((float)pw + 1.0f) * bw) + xs, 0), WW);
        float m = -INFINITY;
        const float* xc = xb + (size_t)c * HH * WW;
        for (int h = hs; h < he; ++h)
            for (int w = ws; w < we; ++w)
                m = fmaxf(m, xc[h * WW + w]);
        float pv = (he > hs && we > ws) ? m : 0.0f;
        split_store(pv, Apk, aswz_idx(r, e, KB1));
    }
}

// ---------------------------------------------------------------------------
// Kernel 2: weight swizzle. B[K,N] f32 -> packed hi/lo fragments in contiguous-k
// 32x16 B layout: frag(nb,kb), lane<16 holds K=kb*32+0..15 for N=nb*16+lane,
// lanes 16..31 hold K=+16..31. One thread per (frag,lane): 16 elems.
// ---------------------------------------------------------------------------
__global__ __launch_bounds__(256)
void swizzleB_kernel(const float* __restrict__ B,
                     ushort_t* __restrict__ Bpk,
                     int K, int N, int NB) {
    int KB = K >> 5;
    int t = blockIdx.x * 256 + threadIdx.x;
    int total = NB * KB * 32;
    if (t >= total) return;
    int lane = t & 31;
    int frag = t >> 5;
    int kb = frag % KB;
    int nb = frag / KB;
    int n  = nb * 16 + (lane & 15);
    int k0 = kb * 32 + ((lane & 16) ? 16 : 0);
    size_t base = (size_t)t * 32;
#pragma unroll
    for (int e = 0; e < 16; ++e) {
        float f = (n < N) ? B[(size_t)(k0 + e) * N + n] : 0.0f;
        split_store(f, Bpk, base + e);
    }
}

// ---------------------------------------------------------------------------
// Kernel 3: register-resident bf16x3 WMMA GEMM. No LDS, no conversions, no
// control flow in the K-loop (EXEC stays all-1s as WMMA requires).
// Block = 8 wave32s as 2(M) x 4(N); wave tile 32x32 = four 16x16 f32 accums;
// 12 v_wmma per K-step vs 8 b128-pair loads from 4 base pointers.
// Out-of-range N-fragments are clamped (duplicate compute, store-guarded).
// ---------------------------------------------------------------------------
#define WMMA_BF16(A, Bm, Cm) \
    __builtin_amdgcn_wmma_f32_16x16x32_bf16(false, (A), false, (Bm), (short)0, (Cm), false, false)

template <bool RELU, bool WF32, bool WSWZ>
__global__ __launch_bounds__(256)
void gemm_wmma_kernel(const ushort_t* __restrict__ Apk,
                      const ushort_t* __restrict__ Bpk,
                      const float* __restrict__ bias,
                      float* __restrict__ Cf,
                      ushort_t* __restrict__ Cpk,
                      int N, int KB, int NB, int KBout) {
    const int tid   = threadIdx.x;
    const int lane  = tid & 31;
    const int wave  = tid >> 5;
    const int waveM = wave >> 2;                 // 0..1
    const int waveN = wave & 3;                  // 0..3
    const int mb0 = blockIdx.y * 4 + waveM * 2;  // two 16-row frags
    const int nb0 = blockIdx.x * 8 + waveN * 2;  // two 16-col frags
    const int nb0c = min(nb0, NB - 1);           // branchless clamp for loads
    const int nb1c = min(nb0 + 1, NB - 1);

    // packed fragment pointers: element stride per kb is 32 lanes * 2 v16bf
    const v16bf* pa0 = (const v16bf*)Apk + ((size_t)mb0 * KB * 32 + lane) * 2;
    const v16bf* pa1 = pa0 + (size_t)KB * 64;
    const v16bf* pb0 = (const v16bf*)Bpk + ((size_t)nb0c * KB * 32 + lane) * 2;
    const v16bf* pb1 = (const v16bf*)Bpk + ((size_t)nb1c * KB * 32 + lane) * 2;

    v8f acc00 = {}, acc01 = {}, acc10 = {}, acc11 = {};

    for (int kb = 0; kb < KB; ++kb) {
        size_t o = (size_t)kb * 64;              // v16bf units per kb step
        v16bf a0h = pa0[o], a0l = pa0[o + 1];
        v16bf a1h = pa1[o], a1l = pa1[o + 1];
        v16bf b0h = pb0[o], b0l = pb0[o + 1];
        v16bf b1h = pb1[o], b1l = pb1[o + 1];
        __builtin_prefetch(&pa0[o + 64], 0, 1);  // speculative: OOB is dropped
        __builtin_prefetch(&pb0[o + 64], 0, 1);
        // round-robin accumulators: no back-to-back same-D WMMA (hazard NOPs)
        acc00 = WMMA_BF16(a0h, b0h, acc00);
        acc10 = WMMA_BF16(a1h, b0h, acc10);
        acc01 = WMMA_BF16(a0h, b1h, acc01);
        acc11 = WMMA_BF16(a1h, b1h, acc11);
        acc00 = WMMA_BF16(a0h, b0l, acc00);
        acc10 = WMMA_BF16(a1h, b0l, acc10);
        acc01 = WMMA_BF16(a0h, b1l, acc01);
        acc11 = WMMA_BF16(a1h, b1l, acc11);
        acc00 = WMMA_BF16(a0l, b0h, acc00);
        acc10 = WMMA_BF16(a1l, b0h, acc10);
        acc01 = WMMA_BF16(a0l, b1h, acc01);
        acc11 = WMMA_BF16(a1l, b1h, acc11);
    }

    // epilogue: bias (+ReLU); optional f32 store and/or re-swizzled store for
    // the next GEMM's A operand. C/D layout: VGPR r = row r (lanes 0-15) or
    // r+8 (lanes 16-31), col = lane&15 within the 16-col frag.
#pragma unroll
    for (int sm = 0; sm < 2; ++sm) {
#pragma unroll
        for (int sn = 0; sn < 2; ++sn) {
            if (nb0 + sn >= NB) continue;        // wave-uniform skip of stores
            v8f acc = (sm == 0) ? ((sn == 0) ? acc00 : acc01)
                                : ((sn == 0) ? acc10 : acc11);
            int col = (nb0 + sn) * 16 + (lane & 15);
            if (col < N) {
                float bv = bias[col];
#pragma unroll
                for (int rr = 0; rr < 8; ++rr) {
                    int row = (mb0 + sm) * 16 + rr + ((lane & 16) ? 8 : 0);
                    float v = acc[rr] + bv;
                    if (RELU) v = fmaxf(v, 0.0f);
                    if (WF32) Cf[(size_t)row * N + col] = v;
                    if (WSWZ) split_store(v, Cpk, aswz_idx(row, col, KBout));
                }
            }
        }
    }
}

// ---------------------------------------------------------------------------
// Launch orchestration.
// d_out layout: [512x84 locs | 512x21 scores | 512x1024 fc7]
// ---------------------------------------------------------------------------
extern "C" void kernel_launch(void* const* d_in, const int* in_sizes, int n_in,
                              void* d_out, int out_size, void* d_ws, size_t ws_size,
                              hipStream_t stream) {
    const float* x    = (const float*)d_in[0];
    const float* rois = (const float*)d_in[1];
    const int*   ridx = (const int*)d_in[2];
    const float* W1   = (const float*)d_in[3];
    const float* b1   = (const float*)d_in[4];
    const float* W2   = (const float*)d_in[5];
    const float* b2   = (const float*)d_in[6];
    const float* Wc   = (const float*)d_in[7];
    const float* bc   = (const float*)d_in[8];
    const float* Wsc  = (const float*)d_in[9];
    const float* bsc  = (const float*)d_in[10];

    float* out    = (float*)d_out;
    float* locs   = out;                              // 512*84
    float* scores = out + (size_t)RR * NLOC;          // 512*21
    float* fc7    = out + (size_t)RR * (NLOC + NCLS); // 512*1024

    // workspace carve-up (ushort units); packed hi/lo = 1024 ushorts per frag
    const size_t szPool = (size_t)(RR / 16) * KB1 * 1024;  // 6.42M
    const size_t szW1   = (size_t)NB_FC * KB1 * 1024;      // 12.8M
    const size_t szW2   = (size_t)NB_FC * KB2 * 1024;      // 2.1M
    const size_t szWc   = (size_t)NB_LOC * KB2 * 1024;
    const size_t szWs   = (size_t)NB_SCR * KB2 * 1024;
    const size_t szAct  = (size_t)(RR / 16) * KB2 * 1024;  // 1.05M

    ushort_t* p = (ushort_t*)d_ws;
    ushort_t* poolP = p; p += szPool;
    ushort_t* w1P = p; p += szW1;
    ushort_t* w2P = p; p += szW2;
    ushort_t* wcP = p; p += szWc;
    ushort_t* wsP = p; p += szWs;
    ushort_t* h1P = p; p += szAct;
    ushort_t* f7P = p; p += szAct;

    // 1) pool -> swizzled A for fc1
    roi_pool_kernel<<<RR, 256, 0, stream>>>(x, rois, ridx, poolP);

    // 2) weight swizzles
    {
        int t1 = NB_FC * KB1 * 32;
        swizzleB_kernel<<<(t1 + 255) / 256, 256, 0, stream>>>(W1, w1P, POOLK, FC, NB_FC);
        int t2 = NB_FC * KB2 * 32;
        swizzleB_kernel<<<(t2 + 255) / 256, 256, 0, stream>>>(W2, w2P, FC, FC, NB_FC);
        int t3 = NB_LOC * KB2 * 32;
        swizzleB_kernel<<<(t3 + 255) / 256, 256, 0, stream>>>(Wc, wcP, FC, NLOC, NB_LOC);
        int t4 = NB_SCR * KB2 * 32;
        swizzleB_kernel<<<(t4 + 255) / 256, 256, 0, stream>>>(Wsc, wsP, FC, NCLS, NB_SCR);
    }

    dim3 blk(256);
    // 3) fc1: relu, swizzled-only output (feeds fc2)
    dim3 g1(NB_FC / 8, RR / 64);  // 8 x 8
    gemm_wmma_kernel<true, false, true><<<g1, blk, 0, stream>>>(
        poolP, w1P, b1, nullptr, h1P, FC, KB1, NB_FC, KB2);

    // 4) fc2: relu, f32 fc7 (output) + swizzled copy (feeds heads)
    dim3 g2(NB_FC / 8, RR / 64);
    gemm_wmma_kernel<true, true, true><<<g2, blk, 0, stream>>>(
        h1P, w2P, b2, fc7, f7P, FC, KB2, NB_FC, KB2);

    // 5) heads: plain f32 outputs
    dim3 g3((NB_LOC + 7) / 8, RR / 64);  // 1 x 8
    gemm_wmma_kernel<false, true, false><<<g3, blk, 0, stream>>>(
        f7P, wcP, bc, locs, nullptr, NLOC, KB2, NB_LOC, 0);

    dim3 g4((NB_SCR + 7) / 8, RR / 64);  // 1 x 8
    gemm_wmma_kernel<false, true, false><<<g4, blk, 0, stream>>>(
        f7P, wsP, bsc, scores, nullptr, NCLS, KB2, NB_SCR, 0);
}
